// MovingAvgGRUModel_56160992363252
// MI455X (gfx1250) — compile-verified, matrix operand order checked by reading, not run on hardware
//
#include <hip/hip_runtime.h>
#include <stdint.h>

// ---------------------------------------------------------------------------
// MovingAvgGRUModel on MI455X (gfx1250, wave32, WMMA)
//
//  * bf16 WMMA (v_wmma_f32_16x16x32_bf16), fp32 accumulation.
//  * recurrent_kernel + kernel pre-packed into WMMA B-fragment layout,
//    staged into LDS once via GLOBAL_LOAD_ASYNC_TO_LDS_B128 (ASYNCcnt path).
//  * hidden state h in registers (lane-exact WMMA C layout); bf16 row-major
//    ping-pong copy of h in L2 feeds next step's A fragments (global_load_b128).
//  * 16 workgroups, per-step sync via CDNA5 cluster barrier (s_barrier -3)
//    with an atomic grid-barrier fallback when not dispatched as a cluster.
//  * global_prefetch of next step's x tile overlaps the recurrent GEMM.
// ---------------------------------------------------------------------------

typedef unsigned short u16;
typedef unsigned int   u32;
typedef __attribute__((ext_vector_type(16))) __bf16 v16bf;
typedef __attribute__((ext_vector_type(8)))  float  v8f;
typedef __attribute__((ext_vector_type(4)))  u32    v4u;

union Frag { v4u q[2]; v16bf v; };

#define BATCH 128
#define SEQ   512
#define FEAT  128
#define HID   512
#define H3    1536
#define D1    64

__device__ __forceinline__ u16 f2bf(float f) {
  u32 x = __float_as_uint(f);
  return (u16)((x + 0x7FFFu + ((x >> 16) & 1u)) >> 16);  // RNE
}
__device__ __forceinline__ float sigm(float x) { return 1.0f / (1.0f + __expf(-x)); }

// ---------------- setup kernels -------------------------------------------

// inputs[b][t][0..127] (stride 129) -> xb[t][b][f] bf16 (t-major, contiguous per step)
__global__ void k_convert_x(const float* __restrict__ in, u16* __restrict__ xb) {
  int tid = blockIdx.x * blockDim.x + threadIdx.x;   // t*16384 + b*128 + f
  int f = tid & 127, b = (tid >> 7) & 127, t = tid >> 14;
  xb[tid] = f2bf(in[(b * SEQ + t) * (FEAT + 1) + f]);
}

// Pack W (K x 1536 row-major fp32) into bf16 WMMA B-fragments.
// Fragment (kt, nt) = 512 u16: [half(2)][lane(32)][8 u16].
// lane l<16: N = nt*16+l, K = kt*32 + {0..7}(half0), {16..23}(half1)
// lane l>=16: same N col (l-16),  K = kt*32 + {8..15}(half0), {24..31}(half1)
__global__ void k_pack(const float* __restrict__ W, u16* __restrict__ out, int nkt) {
  int tid = blockIdx.x * blockDim.x + threadIdx.x;
  int lane = tid & 31;
  int nt   = (tid >> 5) % 96;
  int kt   = tid / (32 * 96);
  if (kt >= nkt) return;
  int n  = nt * 16 + (lane & 15);
  int kb = kt * 32 + (lane >> 4) * 8;
  u16* o = out + (kt * 96 + nt) * 512;
#pragma unroll
  for (int j = 0; j < 8; ++j) o[lane * 8 + j]       = f2bf(W[(kb + j)      * H3 + n]);
#pragma unroll
  for (int j = 0; j < 8; ++j) o[256 + lane * 8 + j] = f2bf(W[(kb + 16 + j) * H3 + n]);
}

__global__ void k_shock(const float* __restrict__ in, float* __restrict__ sm) {
  int b = threadIdx.x;
  float s = 0.f;
  for (int t = 0; t < SEQ; ++t) s += in[(b * SEQ + t) * (FEAT + 1) + FEAT];
  sm[b] = s * (1.0f / (float)SEQ);
}

__global__ void k_zero(u32* p, int n) {
  int i = blockIdx.x * blockDim.x + threadIdx.x;
  if (i < n) p[i] = 0u;
}

// ---------------- GRU scan -------------------------------------------------

__device__ __forceinline__ void grid_barrier(u32* cnt, u32 target) {
  __threadfence();
  __syncthreads();
  if (threadIdx.x == 0) {
    __hip_atomic_fetch_add(cnt, 1u, __ATOMIC_RELAXED, __HIP_MEMORY_SCOPE_AGENT);
    while (__hip_atomic_load(cnt, __ATOMIC_RELAXED, __HIP_MEMORY_SCOPE_AGENT) < target)
      __builtin_amdgcn_s_sleep(1);
  }
  __syncthreads();
  __threadfence();
}

__global__ __launch_bounds__(256) void k_scan(
    const u16* __restrict__ xb, const u16* __restrict__ Rp, const u16* __restrict__ Kp,
    const float* __restrict__ bias, u16* __restrict__ hb0, u16* __restrict__ hb1,
    float* __restrict__ hfin, u32* cnt) {
  // LDS: recurrent B-fragments (16 kt x 6 nt) + input-proj fragments (4 kt x 6 nt)
  __shared__ u16 ldsW[(16 * 6 + 4 * 6) * 512];  // 122880 B
  u16* ldsR = ldsW;
  u16* ldsK = ldsW + 16 * 6 * 512;

  const int g = blockIdx.x;            // 0..15 : owns h columns [g*32, g*32+32)
  const int tid = threadIdx.x;
  const int w = tid >> 5;              // wave id = M tile (rows w*16..w*16+15)
  const int lid = tid & 31;
  const int lrow = lid & 15;
  const int khalf = lid >> 4;

  int ntg[6];                          // global N-tiles: gate c, subtile s
#pragma unroll
  for (int c = 0; c < 3; ++c)
#pragma unroll
    for (int s = 0; s < 2; ++s) ntg[c * 2 + s] = c * 32 + g * 2 + s;

  // ---- Stage this WG's weight fragments into LDS via the CDNA5 async path:
  //      GLOBAL_LOAD_ASYNC_TO_LDS_B128 (memory -> LDS, tracked by ASYNCcnt). ----
  {
    const u32 lds_base = (u32)(uintptr_t)(&ldsW[0]);  // addr[31:0] == LDS offset
    for (int chunk = tid; chunk < (16 * 6 + 4 * 6) * 64; chunk += 256) {
      int blk = chunk >> 6;            // 1KB fragment block
      int off = (chunk & 63) * 16;     // byte offset within block
      const u16* src;
      if (blk < 96) {
        int kt = blk / 6, f = blk - kt * 6;
        src = Rp + ((kt * 96 + ntg[f]) << 9);
      } else {
        int b2 = blk - 96;
        int kt = b2 / 6, f = b2 - kt * 6;
        src = Kp + ((kt * 96 + ntg[f]) << 9);
      }
      u32 lds_off = lds_base + (u32)(chunk * 16);
      unsigned long long gaddr = (unsigned long long)(uintptr_t)src + (unsigned long long)off;
      asm volatile("global_load_async_to_lds_b128 %0, %1, off"
                   :: "v"(lds_off), "v"(gaddr)
                   : "memory");
    }
    asm volatile("s_wait_asynccnt 0" ::: "memory");
  }
  __syncthreads();

  // Per-lane gate biases (column = g*32 + s*16 + lrow), loop-invariant.
  float bz[2], br_[2], bx[2], bh[2];
#pragma unroll
  for (int s = 0; s < 2; ++s) {
    int n = g * 32 + s * 16 + lrow;
    bz[s]  = bias[n]        + bias[H3 + n];          // z: b_i + b_r
    br_[s] = bias[HID + n]  + bias[H3 + HID + n];    // r: b_i + b_r
    bx[s]  = bias[2 * HID + n];                      // xh: b_i
    bh[s]  = bias[H3 + 2 * HID + n];                 // hh_: b_r
  }

  // Hidden-state master copy in registers: hm[s*8+j] = h[row w*16+khalf*8+j][col]
  float hm[16];
#pragma unroll
  for (int i = 0; i < 16; ++i) hm[i] = 0.f;

  // Ping-pong pointers kept as a phi of two known-global pointers so the
  // backend emits global_load_b128 / global_store_b16 (not FLAT).
  u16* hcur = hb0;
  u16* hnxt = hb1;
  const bool clustered = (__builtin_amdgcn_cluster_id_x() != 0);

  v8f vzero;
#pragma unroll
  for (int j = 0; j < 8; ++j) vzero[j] = 0.f;

  for (int t = 0; t < SEQ; ++t) {
    // Prefetch next step's x rows for this wave (global_prefetch_b8) so the
    // fetch overlaps the recurrent GEMM below.
    if (t + 1 < SEQ) {
      const u16* nx = xb + (t + 1) * (BATCH * FEAT) + (w * 16 + lrow) * FEAT + khalf * 64;
      __builtin_prefetch(nx, 0, 1);
    }

    // acc[0..1]=z(s=0,1)  acc[2..3]=r  acc[4..5]=xh  acc[6..7]=hh_
    v8f acc[8];
#pragma unroll
    for (int i = 0; i < 8; ++i) acc[i] = vzero;

    // ---- recurrent: h(128x512) @ R  -> 6 N-tiles, K=512 ----
    for (int kt = 0; kt < 16; ++kt) {
      Frag a;
      const u16* ap = hcur + (w * 16 + lrow) * HID + kt * 32 + khalf * 8;
      a.q[0] = *(const v4u*)ap;
      a.q[1] = *(const v4u*)(ap + 16);
#pragma unroll
      for (int f = 0; f < 6; ++f) {
        Frag b;
        const u16* bp = ldsR + (kt * 6 + f) * 512 + lid * 8;
        b.q[0] = *(const v4u*)bp;
        b.q[1] = *(const v4u*)(bp + 256);
        int ai = (f < 4) ? f : f + 2;   // gate-2 recurrent part kept separate
        acc[ai] = __builtin_amdgcn_wmma_f32_16x16x32_bf16(
            false, a.v, false, b.v, (short)0, acc[ai], false, false);
      }
    }

    // ---- input projection: x_t(128x128) @ K -> same tiles, K=128 ----
    const u16* xt = xb + t * (BATCH * FEAT);
#pragma unroll
    for (int kt = 0; kt < 4; ++kt) {
      Frag a;
      const u16* ap = xt + (w * 16 + lrow) * FEAT + kt * 32 + khalf * 8;
      a.q[0] = *(const v4u*)ap;
      a.q[1] = *(const v4u*)(ap + 16);
#pragma unroll
      for (int f = 0; f < 6; ++f) {
        Frag b;
        const u16* bp = ldsK + (kt * 6 + f) * 512 + lid * 8;
        b.q[0] = *(const v4u*)bp;
        b.q[1] = *(const v4u*)(bp + 256);
        acc[f] = __builtin_amdgcn_wmma_f32_16x16x32_bf16(
            false, a.v, false, b.v, (short)0, acc[f], false, false);
      }
    }

    // ---- gates + state update (lane-exact WMMA C layout) ----
#pragma unroll
    for (int s = 0; s < 2; ++s) {
      int ncol = g * 32 + s * 16 + lrow;
#pragma unroll
      for (int j = 0; j < 8; ++j) {
        float z = sigm(acc[s][j] + bz[s]);
        float r = sigm(acc[2 + s][j] + br_[s]);
        float cand = tanhf(acc[4 + s][j] + bx[s] + r * (acc[6 + s][j] + bh[s]));
        float hnew = z * hm[s * 8 + j] + (1.0f - z) * cand;
        hm[s * 8 + j] = hnew;
        hnxt[(w * 16 + khalf * 8 + j) * HID + ncol] = f2bf(hnew);
      }
    }

    if (t != SEQ - 1) {
      if (clustered) {
        __threadfence();
        __syncthreads();
        __builtin_amdgcn_s_cluster_barrier();   // s_barrier_signal/-wait -3
        __threadfence();
      } else {
        grid_barrier(cnt, 16u * (u32)(t + 1));
      }
    }

    u16* tsw = hcur; hcur = hnxt; hnxt = tsw;   // ping-pong swap (phi of globals)
  }

  // Final state (fp32) for the head.
#pragma unroll
  for (int s = 0; s < 2; ++s)
#pragma unroll
    for (int j = 0; j < 8; ++j)
      hfin[(w * 16 + khalf * 8 + j) * HID + g * 32 + s * 16 + lrow] = hm[s * 8 + j];
}

// ---------------- head -----------------------------------------------------

__global__ void k_head(const float* __restrict__ hfin, const float* __restrict__ sm,
                       const float* __restrict__ W1, const float* __restrict__ b1,
                       const float* __restrict__ gamma, const float* __restrict__ beta,
                       const float* __restrict__ mm, const float* __restrict__ mv,
                       const float* __restrict__ W2, const float* __restrict__ b2,
                       const float* __restrict__ Wsh, float* __restrict__ out) {
  __shared__ float red[D1];
  int b = blockIdx.x, j = threadIdx.x;
  float add = sm[b] * Wsh[0];
  float acc = b1[j];
  for (int k = 0; k < HID; ++k) acc += (hfin[b * HID + k] + add) * W1[k * D1 + j];
  float h = fmaxf(acc, 0.f);
  h = (h - mm[j]) * rsqrtf(mv[j] + 0.001f) * gamma[j] + beta[j];
  red[j] = h * W2[j];
  __syncthreads();
  for (int off = 32; off > 0; off >>= 1) {
    if (j < off) red[j] += red[j + off];
    __syncthreads();
  }
  if (j == 0) out[b] = red[0] + b2[0];
}

// ---------------- launch ----------------------------------------------------

extern "C" void kernel_launch(void* const* d_in, const int* in_sizes, int n_in,
                              void* d_out, int out_size, void* d_ws, size_t ws_size,
                              hipStream_t stream) {
  const float* in_x   = (const float*)d_in[0];
  const float* kern   = (const float*)d_in[1];
  const float* rkern  = (const float*)d_in[2];
  const float* bias   = (const float*)d_in[3];
  const float* W1     = (const float*)d_in[4];
  const float* b1     = (const float*)d_in[5];
  const float* gamma  = (const float*)d_in[6];
  const float* beta   = (const float*)d_in[7];
  const float* mmean  = (const float*)d_in[8];
  const float* mvar   = (const float*)d_in[9];
  const float* W2     = (const float*)d_in[10];
  const float* b2     = (const float*)d_in[11];
  const float* Wsh    = (const float*)d_in[12];
  float* out = (float*)d_out;

  char* ws = (char*)d_ws;
  u16*  xb   = (u16*)ws;                    ws += (size_t)SEQ * BATCH * FEAT * 2;  // 16 MB
  u16*  Rp   = (u16*)ws;                    ws += (size_t)16 * 96 * 512 * 2;       // 1.5 MB
  u16*  Kp   = (u16*)ws;                    ws += (size_t)4 * 96 * 512 * 2;        // 384 KB
  u16*  hb0  = (u16*)ws;                    ws += (size_t)BATCH * HID * 2;
  u16*  hb1  = (u16*)ws;                    ws += (size_t)BATCH * HID * 2;
  float* hfin = (float*)ws;                 ws += (size_t)BATCH * HID * 4;
  float* sm   = (float*)ws;                 ws += 512;
  u32*   cnt  = (u32*)ws;                   ws += 256;

  // setup
  k_convert_x<<<(SEQ * BATCH * FEAT) / 256, 256, 0, stream>>>(in_x, xb);
  k_pack<<<(16 * 96 * 32) / 256, 256, 0, stream>>>(rkern, Rp, 16);
  k_pack<<<(4 * 96 * 32) / 256, 256, 0, stream>>>(kern, Kp, 4);
  k_shock<<<1, 128, 0, stream>>>(in_x, sm);
  // zero hb0|hb1|hfin|sm|cnt (contiguous region)
  int zwords = (BATCH * HID * 2 * 2 + BATCH * HID * 4 + 512 + 256) / 4;
  k_zero<<<(zwords + 255) / 256, 256, 0, stream>>>((u32*)hb0, zwords);

  // 512-step GRU scan: 16 WGs (one cluster's worth), 8 waves each
  k_scan<<<16, 256, 0, stream>>>(xb, Rp, Kp, bias, hb0, hb1, hfin, cnt);

  // head: relu(state@W1+b1) -> BN -> @W2+b2
  k_head<<<BATCH, D1, 0, stream>>>(hfin, sm, W1, b1, gamma, beta, mmean, mvar,
                                   W2, b2, Wsh, out);
}
